// KoopmanOperator_72773925864294
// MI455X (gfx1250) — matrix operand order repaired and checked by reading, not compiled, not run on hardware
//
#include <hip/hip_runtime.h>

typedef __attribute__((ext_vector_type(16))) _Float16 v16h;
typedef __attribute__((ext_vector_type(8)))  _Float16 v8h;
typedef __attribute__((ext_vector_type(8)))  float    v8f;

namespace {
constexpr int RR = 4;     // real modes
constexpr int PP = 6;     // complex modes
constexpr int HH = 64;    // hidden
constexpr int CH = 16;    // channels = R + 2P
constexpr int THREADS = 128;
constexpr int WAVES = 4;
constexpr int ROWS_PER_BLOCK = WAVES * 16;
}

// Combine two 8-half vectors into one 16-half WMMA fragment register block.
static __device__ inline v16h combine8(v8h lo, v8h hi) {
  v16h r;
#pragma unroll
  for (int e = 0; e < 8; ++e) { r[e] = lo[e]; r[8 + e] = hi[e]; }
  return r;
}

// B fragment: two 8-f16 runs at p and p+16 (K {0..7,16..23}+half*8) from transposed-weight LDS.
static __device__ inline v16h load_b_frag(const _Float16* p) {
  return combine8(*(const v8h*)(p), *(const v8h*)(p + 16));
}

__global__ __launch_bounds__(THREADS) void koopman_wmma_kernel(
    const float* __restrict__ z,
    const float* __restrict__ W1r, const float* __restrict__ b1r,
    const float* __restrict__ Wmr, const float* __restrict__ bmr,
    const float* __restrict__ Wfr, const float* __restrict__ bfr,
    const float* __restrict__ W1c, const float* __restrict__ b1c,
    const float* __restrict__ Wmc, const float* __restrict__ bmc,
    const float* __restrict__ Wfc, const float* __restrict__ bfc,
    float* __restrict__ out)
{
  // Transposed f16 weights: sWT[n*64 + h] = W[h][n]  (B fragment = contiguous K per lane)
  __shared__ __align__(16) _Float16 sWT1[HH * HH];
  __shared__ __align__(16) _Float16 sWT2[HH * HH];
  // Per-wave layer-1 activation tile, row-major [row][k], f16
  __shared__ __align__(16) _Float16 sH[WAVES][16 * HH];
  __shared__ float sW1[HH], sB1[HH], sBm1[HH], sBm2[HH], sWf0[HH], sWf1[HH];
  __shared__ float sRed[WAVES][32];

  const int tid  = threadIdx.x;
  const int wave = tid >> 5;
  const int lane = tid & 31;
  const int half = lane >> 4;   // K-half of the wave for A/B fragments
  const int nn   = lane & 15;   // row (A) / column (B,C,D) index

  const int m = blockIdx.y;
  const bool isReal = (m < RR);
  const int mode = isReal ? m : (m - RR);

  const float* Wm0 = isReal ? (Wmr + (0 * RR + mode) * HH * HH)
                            : (Wmc + (0 * PP + mode) * HH * HH);
  const float* Wm1 = isReal ? (Wmr + (1 * RR + mode) * HH * HH)
                            : (Wmc + (1 * PP + mode) * HH * HH);

  // ---- Stage this mode's weights to LDS (coalesced global reads, f32->f16, transpose) ----
  for (int i = tid; i < HH * HH; i += THREADS) {
    const int h = i >> 6;
    const int c = i & 63;
    sWT1[(c << 6) | h] = (_Float16)Wm0[i];
    sWT2[(c << 6) | h] = (_Float16)Wm1[i];
  }
  if (tid < HH) {
    sW1[tid]  = isReal ? W1r[mode * HH + tid] : W1c[mode * HH + tid];
    sB1[tid]  = isReal ? b1r[mode * HH + tid] : b1c[mode * HH + tid];
    sBm1[tid] = isReal ? bmr[(0 * RR + mode) * HH + tid] : bmc[(0 * PP + mode) * HH + tid];
    sBm2[tid] = isReal ? bmr[(1 * RR + mode) * HH + tid] : bmc[(1 * PP + mode) * HH + tid];
    sWf0[tid] = isReal ? Wfr[mode * HH + tid] : Wfc[(mode * HH + tid) * 2 + 0];
    sWf1[tid] = isReal ? 0.0f               : Wfc[(mode * HH + tid) * 2 + 1];
  }
  __syncthreads();

  const int row0 = blockIdx.x * ROWS_PER_BLOCK + wave * 16;
  const int row  = row0 + nn;   // this lane's sample row (same for lane and lane+16)

  // ---- Layer 0: scalar lift, generated directly in WMMA A-fragment layout ----
  float x;
  if (isReal) {
    x = z[row * CH + mode];
  } else {
    const float c0 = z[row * CH + RR + 2 * mode + 0];
    const float c1 = z[row * CH + RR + 2 * mode + 1];
    x = c0 * c0 + c1 * c1;
  }

  v16h a0, a1;  // K = 0..31 and 32..63 fragments
#pragma unroll
  for (int e = 0; e < 16; ++e) {
    // A 16x32 f16 layout: lanes<16 hold K {0..7,16..23}(+half*8), lanes>=16 K {8..15,24..31}
    const int kb = (e < 8) ? (half * 8 + e) : (16 + half * 8 + (e - 8));
    a0[e] = (_Float16)fmaxf(fmaf(x, sW1[kb],      sB1[kb]),      0.0f);
    a1[e] = (_Float16)fmaxf(fmaf(x, sW1[kb + 32], sB1[kb + 32]), 0.0f);
  }

  // ---- Layer 1: 16x64 @ 64x64, bias folded into C init ----
  v8f acc[4];
#pragma unroll
  for (int t = 0; t < 4; ++t) {
    const int col = t * 16 + nn;
    const _Float16* pb = sWT1 + col * HH + half * 8;
    const v16h bk0 = load_b_frag(pb);        // K = 0..31 (this lane's half)
    const v16h bk1 = load_b_frag(pb + 32);   // K = 32..63
    const float bias = sBm1[col];
    v8f c;
#pragma unroll
    for (int j = 0; j < 8; ++j) c[j] = bias;
    c = __builtin_amdgcn_wmma_f32_16x16x32_f16(false, a0, false, bk0, (short)0, c, false, false);
    c = __builtin_amdgcn_wmma_f32_16x16x32_f16(false, a1, false, bk1, (short)0, c, false, false);
    acc[t] = c;
  }

  // relu, spill to wave-private LDS tile as f16 (D layout -> row-major)
#pragma unroll
  for (int t = 0; t < 4; ++t) {
#pragma unroll
    for (int j = 0; j < 8; ++j) {
      sH[wave][(half * 8 + j) * HH + t * 16 + nn] = (_Float16)fmaxf(acc[t][j], 0.0f);
    }
  }
  __syncthreads();

  // ---- Layer 2: reload A fragments (aligned ds_load_b128s) ----
  const _Float16* ph = sH[wave] + nn * HH;
  const v16h a2_0 = combine8(*(const v8h*)(ph + half * 8),      *(const v8h*)(ph + 16 + half * 8));
  const v16h a2_1 = combine8(*(const v8h*)(ph + 32 + half * 8), *(const v8h*)(ph + 48 + half * 8));

  v8f acc2[4];
#pragma unroll
  for (int t = 0; t < 4; ++t) {
    const int col = t * 16 + nn;
    const _Float16* pb = sWT2 + col * HH + half * 8;
    const v16h bk0 = load_b_frag(pb);
    const v16h bk1 = load_b_frag(pb + 32);
    const float bias = sBm2[col];
    v8f c;
#pragma unroll
    for (int j = 0; j < 8; ++j) c[j] = bias;
    c = __builtin_amdgcn_wmma_f32_16x16x32_f16(false, a2_0, false, bk0, (short)0, c, false, false);
    c = __builtin_amdgcn_wmma_f32_16x16x32_f16(false, a2_1, false, bk1, (short)0, c, false, false);
    acc2[t] = c;
  }

  // ---- Head: per-row dot(s) over 64 columns (relu applied), cross-lane reduce ----
  float pm[8], po[8];
#pragma unroll
  for (int j = 0; j < 8; ++j) { pm[j] = 0.0f; po[j] = 0.0f; }
#pragma unroll
  for (int t = 0; t < 4; ++t) {
    const float wf0 = sWf0[t * 16 + nn];
    const float wf1 = sWf1[t * 16 + nn];
#pragma unroll
    for (int j = 0; j < 8; ++j) {
      const float hv = fmaxf(acc2[t][j], 0.0f);
      pm[j] = fmaf(hv, wf0, pm[j]);
      po[j] = fmaf(hv, wf1, po[j]);
    }
  }
#pragma unroll
  for (int j = 0; j < 8; ++j) {
#pragma unroll
    for (int mask = 1; mask < 16; mask <<= 1) {
      pm[j] += __shfl_xor(pm[j], mask, 32);
      po[j] += __shfl_xor(po[j], mask, 32);
    }
  }
  if (nn == 0) {
#pragma unroll
    for (int j = 0; j < 8; ++j) {
      sRed[wave][half * 8 + j]      = pm[j];   // rows 0..7 from lane 0, 8..15 from lane 16
      sRed[wave][16 + half * 8 + j] = po[j];
    }
  }
  __syncthreads();

  // ---- Elementwise epilogue: one row per lane (lanes 0..15) ----
  if (half == 0) {
    if (isReal) {
      const float lam = sRed[wave][nn] + bfr[mode];
      const int idx = row * CH + mode;
      out[idx] = z[idx] * lam;
    } else {
      const float mu = sRed[wave][nn]      + bfc[mode * 2 + 0];
      const float om = sRed[wave][16 + nn] + bfc[mode * 2 + 1];
      const float sc = __expf(mu);
      const float cs = __cosf(om);
      const float sn = __sinf(om);
      const int base = row * CH + RR + 2 * mode;
      const float c0 = z[base + 0];
      const float c1 = z[base + 1];
      out[base + 0] = sc * ( c0 * cs + c1 * sn);
      out[base + 1] = sc * (-c0 * sn + c1 * cs);
    }
  }
}

extern "C" void kernel_launch(void* const* d_in, const int* in_sizes, int n_in,
                              void* d_out, int out_size, void* d_ws, size_t ws_size,
                              hipStream_t stream) {
  const float* z   = (const float*)d_in[0];
  const float* W1r = (const float*)d_in[1];
  const float* b1r = (const float*)d_in[2];
  const float* Wmr = (const float*)d_in[3];
  const float* bmr = (const float*)d_in[4];
  const float* Wfr = (const float*)d_in[5];
  const float* bfr = (const float*)d_in[6];
  const float* W1c = (const float*)d_in[7];
  const float* b1c = (const float*)d_in[8];
  const float* Wmc = (const float*)d_in[9];
  const float* bmc = (const float*)d_in[10];
  const float* Wfc = (const float*)d_in[11];
  const float* bfc = (const float*)d_in[12];
  float* out = (float*)d_out;

  const int nrows = in_sizes[0] / CH;          // B*S = 131072
  dim3 grid(nrows / ROWS_PER_BLOCK, RR + PP);  // (2048, 10)
  koopman_wmma_kernel<<<grid, THREADS, 0, stream>>>(
      z, W1r, b1r, Wmr, bmr, Wfr, bfr, W1c, b1c, Wmc, bmc, Wfc, bfc, out);
}
